// NER_85461259256082
// MI455X (gfx1250) — matrix operand order repaired and checked by reading, not compiled, not run on hardware
//
#include <hip/hip_runtime.h>
#include <hip/hip_bf16.h>

// ---------------------------------------------------------------------------
// Problem dims (fixed by reference): B=16, L=512, D=1280, H=512, C=5, Tp=510
// ---------------------------------------------------------------------------
#define BB   16
#define LL   512
#define DD   1280
#define HH   512
#define CC   5
#define TP   510            // L-2
#define ROWS (BB * TP)      // 8160
#define G4H  (4 * HH)       // 2048

typedef __attribute__((ext_vector_type(16))) __bf16       v16bf;
typedef __attribute__((ext_vector_type(8)))  float        v8f;
typedef __attribute__((ext_vector_type(8)))  unsigned int v8u;
typedef __attribute__((ext_vector_type(4)))  unsigned int v4u;
typedef __attribute__((ext_vector_type(8)))  int          v8i;
typedef __attribute__((ext_vector_type(4)))  int          v4i;

#ifndef __has_builtin
#define __has_builtin(x) 0
#endif
#if __has_builtin(__builtin_amdgcn_tensor_load_to_lds) && \
    __has_builtin(__builtin_amdgcn_s_wait_tensorcnt)
#define HAVE_TDM 1
#else
#define HAVE_TDM 0
#endif

__device__ __forceinline__ unsigned short f32_to_bf16(float f) {
    unsigned int u = __builtin_bit_cast(unsigned int, f);
    unsigned int r = (u + 0x7FFFu + ((u >> 16) & 1u)) >> 16;  // RNE
    return (unsigned short)r;
}

// A fragment (16x32 bf16, M x K). Source is row-major [M][K] bf16, K-contig.
// Lane l: M = l%16, K = kbase + (l/16)*8 + {0..7} in v0..3, +16 in v4..7.
__device__ __forceinline__ v16bf load_frag_a(const unsigned int* base_dw,
                                             int ld_dw, int kbase, int lane) {
    int m    = lane & 15;
    int half = lane >> 4;
    const unsigned int* p = base_dw + (size_t)m * ld_dw + (kbase >> 1) + half * 4;
    v8u u;
    u[0] = p[0]; u[1] = p[1]; u[2] = p[2]; u[3] = p[3];
    p += 8;  // +16 shorts
    u[4] = p[0]; u[5] = p[1]; u[6] = p[2]; u[7] = p[3];
    return __builtin_bit_cast(v16bf, u);
}

// B fragment (32x16 bf16, K x N). Source is row-major weight [N][K] bf16.
// Lane l: N = ntile + l%16, K = kbase + (l/16)*8 + {0..7} in v0..3, +16 in v4..7.
__device__ __forceinline__ v16bf load_frag_b(const unsigned int* mat_dw,
                                             int ld_dw, int ntile, int kbase,
                                             int lane) {
    int n    = ntile + (lane & 15);
    int half = lane >> 4;
    const unsigned int* p = mat_dw + (size_t)n * ld_dw + (kbase >> 1) + half * 4;
    v8u u;
#pragma unroll
    for (int i = 0; i < 8; ++i) u[i] = p[(i & 3) + (i >> 2) * 8];
    return __builtin_bit_cast(v16bf, u);
}

// ---------------------------------------------------------------------------
// 1) lengths-2 from attention mask
// ---------------------------------------------------------------------------
__global__ void k_lengths(const int* __restrict__ mask, int* __restrict__ len2) {
    __shared__ int s[256];
    int b = blockIdx.x, t = threadIdx.x;
    s[t] = mask[b * LL + t] + mask[b * LL + 256 + t];
    __syncthreads();
    for (int w = 128; w > 0; w >>= 1) {
        if (t < w) s[t] += s[t + w];
        __syncthreads();
    }
    if (t == 0) len2[b] = s[0] - 2;
}

// ---------------------------------------------------------------------------
// 2) pack x = masked embeddings[:,1:-1] -> bf16 (ROWS x D)
// ---------------------------------------------------------------------------
__global__ void k_pack(const float* __restrict__ emb, const int* __restrict__ len2,
                       unsigned short* __restrict__ xb) {
    long long gid = (long long)blockIdx.x * 256 + threadIdx.x;
    if (gid >= (long long)ROWS * DD) return;
    int row = (int)(gid / DD);
    int d   = (int)(gid % DD);
    int b = row / TP, t = row % TP;
    float v = (t < len2[b]) ? emb[((size_t)b * LL + (t + 1)) * DD + d] : 0.f;
    xb[gid] = f32_to_bf16(v);
}

// ---------------------------------------------------------------------------
// 3) f32 -> bf16 weight conversion
// ---------------------------------------------------------------------------
__global__ void k_cvt_bf16(const float* __restrict__ in,
                           unsigned short* __restrict__ out, int n) {
    int i = blockIdx.x * 256 + threadIdx.x;
    if (i < n) out[i] = f32_to_bf16(in[i]);
}

__global__ void k_bias(const float* __restrict__ bihf, const float* __restrict__ bhhf,
                       const float* __restrict__ bihb, const float* __restrict__ bhhb,
                       float* __restrict__ bias) {
    int i = blockIdx.x * 256 + threadIdx.x;
    if (i < G4H) bias[i] = bihf[i] + bhhf[i];
    else if (i < 2 * G4H) bias[i] = bihb[i - G4H] + bhhb[i - G4H];
}

// ---------------------------------------------------------------------------
// 4) xg = x @ w_ih^T + (b_ih + b_hh).  WMMA bf16.
//    A tile (16 x 1280 bf16, 40KB) staged to LDS via TDM tensor_load_to_lds
//    when available (TENSORcnt), else cooperative copy.  Each wave computes
//    4 N-tiles so the A fragment is reused 4x per K-step.
//    grid (510 row-tiles, 4 N-supertiles of 512, 2 dirs), 256 thr = 8 waves.
// ---------------------------------------------------------------------------
__global__ void k_xg_gemm(const unsigned short* __restrict__ xb,
                          const unsigned short* __restrict__ wih_bf,
                          const float* __restrict__ bias,
                          float* __restrict__ xg) {
    extern __shared__ unsigned char smem[];
    unsigned int* aT = (unsigned int*)smem;   // 16 rows x 640 dwords (bf16 pairs)
    const int LD = DD / 2;                    // 640

    int tid = threadIdx.x;
    int rowTile = blockIdx.x;
    int dir = blockIdx.z;

    const unsigned short* srcRow = xb + (size_t)rowTile * 16 * DD;

#if HAVE_TDM
    // One TDM descriptor moves the whole 16x1280 bf16 tile global->LDS.
    if (tid == 0) {
        unsigned lds_off = (unsigned)(size_t)(void*)aT;     // LDS aperture: low 32 bits
        unsigned long long ga = (unsigned long long)(size_t)srcRow;
        v4u g0;
        g0[0] = 1u;                                          // count=1, user mode
        g0[1] = lds_off;                                     // lds_addr
        g0[2] = (unsigned)(ga & 0xFFFFFFFFull);              // global_addr[31:0]
        g0[3] = (unsigned)((ga >> 32) & 0x01FFFFFFull)       // global_addr[56:32]
              | 0x80000000u;                                 // type=2 ("image")
        v8i g1;
        g1[0] = 0x00010000;                                  // data_size=1 (2B), mask=0
        g1[1] = (int)(((unsigned)DD & 0xFFFFu) << 16);       // tensor_dim0[15:0]
        g1[2] = (int)(((unsigned)DD >> 16) | (16u << 16));   // tensor_dim0 hi | tensor_dim1 lo
        g1[3] = (int)(((unsigned)DD) << 16);                 // tensor_dim1 hi | tile_dim0=1280
        g1[4] = 16;                                          // tile_dim1=16, tile_dim2=0
        g1[5] = DD;                                          // tensor_dim0_stride lo
        g1[6] = 0;                                           // stride hi | dim1_stride lo
        g1[7] = 0;
        v4i z4 = {};
        v8i z8 = {};
#if __clang_major__ >= 23
        __builtin_amdgcn_tensor_load_to_lds(g0, g1, z4, z4, z8, 0);
#else
        __builtin_amdgcn_tensor_load_to_lds(g0, g1, z4, z4, 0);
#endif
        __builtin_amdgcn_s_wait_tensorcnt(0);
    }
    __syncthreads();
#else
    {
        const unsigned int* src = (const unsigned int*)srcRow;
        for (int i = tid; i < 16 * LD; i += 256) aT[i] = src[i];
        __syncthreads();
    }
#endif

    int wave = tid >> 5, lane = tid & 31;
    int nbase = blockIdx.y * 512 + wave * 64;     // 4 N-tiles of 16 per wave
    const unsigned int* w32 =
        (const unsigned int*)wih_bf + (size_t)dir * G4H * LD;

    v8f acc[4] = {};
    for (int kk = 0; kk < DD / 32; ++kk) {
        v16bf a = load_frag_a(aT, LD, kk * 32, lane);
#pragma unroll
        for (int tI = 0; tI < 4; ++tI) {
            v16bf b = load_frag_b(w32, LD, nbase + tI * 16, kk * 32, lane);
            acc[tI] = __builtin_amdgcn_wmma_f32_16x16x32_bf16(
                false, a, false, b, (short)0, acc[tI], false, false);
        }
    }

    int half = lane >> 4, nl = lane & 15;
    float* out = xg + (size_t)dir * ROWS * G4H;
#pragma unroll
    for (int tI = 0; tI < 4; ++tI) {
        int n = nbase + tI * 16 + nl;
        float bv = bias[dir * G4H + n];
#pragma unroll
        for (int r = 0; r < 8; ++r) {
            int M = r + 8 * half;
            int row = rowTile * 16 + M;
            out[(size_t)row * G4H + n] = acc[tI][r] + bv;
        }
    }
}

// ---------------------------------------------------------------------------
// 5) Sequential LSTM, one workgroup per direction (grid.x = 2), 1024 threads.
//    Per step: g = xg_t + h @ w_hh^T via WMMA (h as bf16 A-operand in LDS),
//    gates in f32, h written back to LDS (bf16) and global (f32).
//    LDS: gL f32[16][2048] = 128KB, hA bf16[16][512] = 16KB (CDNA5: 320KB/WGP)
//    Next-step xg slice prefetched (global_prefetch_b8) to hide the only
//    per-step HBM/L2 latency on the 510-deep dependence chain.
// ---------------------------------------------------------------------------
__global__ void __launch_bounds__(1024) k_lstm(const float* __restrict__ xg_all,
                                               const unsigned short* __restrict__ whh_bf,
                                               const float* __restrict__ h0,
                                               const float* __restrict__ c0,
                                               float* __restrict__ hout_all) {
    extern __shared__ unsigned char smem[];
    float* gL = (float*)smem;                                   // 16*2048 f32
    unsigned short* hA = (unsigned short*)(smem + 16 * G4H * 4); // 16*512 bf16
    unsigned int* hA32 = (unsigned int*)hA;
    const int LDH = HH / 2;  // 256 dwords per hidden row

    int dir = blockIdx.x;
    int tid = threadIdx.x;
    const float* xg = xg_all + (size_t)dir * ROWS * G4H;
    const unsigned int* w32 = (const unsigned int*)whh_bf + (size_t)dir * G4H * LDH;
    float* hout = hout_all + (size_t)dir * ROWS * HH;

    // init state: each thread owns 8 (b, j) cells
    float creg[8];
#pragma unroll
    for (int r = 0; r < 8; ++r) {
        int idx = tid + r * 1024;
        creg[r] = c0[dir * (BB * HH) + idx];
        hA[idx] = f32_to_bf16(h0[dir * (BB * HH) + idx]);
    }

    int wave = tid >> 5, lane = tid & 31;
    int nbase = wave * 64;                 // 4 N-tiles of 16 per wave
    int half = lane >> 4, nl = lane & 15;

    for (int s = 0; s < TP; ++s) {
        int tt = dir ? (TP - 1 - s) : s;
        __syncthreads();  // hA ready

        // prefetch next step's xg slice for this wave's N-range
        if (s + 1 < TP) {
            int tt2 = dir ? (TP - 2 - s) : (s + 1);
            int prow = lane & 15;                 // batch
            int pn   = nbase + (lane >> 4) * 32;  // 2 chunks of 32 floats
            __builtin_prefetch(&xg[((size_t)prow * TP + tt2) * G4H + pn], 0, 1);
        }

        v8f acc[4] = {};
        for (int kk = 0; kk < HH / 32; ++kk) {
            v16bf a = load_frag_a(hA32, LDH, kk * 32, lane);
#pragma unroll
            for (int tI = 0; tI < 4; ++tI) {
                v16bf b = load_frag_b(w32, LDH, nbase + tI * 16, kk * 32, lane);
                acc[tI] = __builtin_amdgcn_wmma_f32_16x16x32_bf16(
                    false, a, false, b, (short)0, acc[tI], false, false);
            }
        }
#pragma unroll
        for (int tI = 0; tI < 4; ++tI) {
            int n = nbase + tI * 16 + nl;
#pragma unroll
            for (int r = 0; r < 8; ++r) {
                int M = r + 8 * half;  // batch row
                float xv = xg[((size_t)M * TP + tt) * G4H + n];
                gL[M * G4H + n] = acc[tI][r] + xv;
            }
        }
        __syncthreads();  // g ready

        // gate update
#pragma unroll
        for (int r = 0; r < 8; ++r) {
            int idx = tid + r * 1024;
            int b = idx >> 9, j = idx & (HH - 1);
            float gi = gL[b * G4H + j];
            float gf = gL[b * G4H + HH + j];
            float gg = gL[b * G4H + 2 * HH + j];
            float go = gL[b * G4H + 3 * HH + j];
            float i_ = 1.f / (1.f + __expf(-gi));
            float f_ = 1.f / (1.f + __expf(-gf));
            float o_ = 1.f / (1.f + __expf(-go));
            float c = f_ * creg[r] + i_ * tanhf(gg);
            creg[r] = c;
            float h = o_ * tanhf(c);
            hout[((size_t)b * TP + tt) * HH + j] = h;
            hA[b * HH + j] = f32_to_bf16(h);
        }
        // next-iteration top barrier orders hA writes vs reads
    }
}

// ---------------------------------------------------------------------------
// 6) emissions = [hf, hb] @ w_out + b_out   (N=5 head, plain VALU)
// ---------------------------------------------------------------------------
__global__ void k_emissions(const float* __restrict__ hout,
                            const float* __restrict__ w_out,
                            const float* __restrict__ b_out,
                            float* __restrict__ em) {
    int gid = blockIdx.x * 256 + threadIdx.x;
    if (gid >= ROWS * CC) return;
    int row = gid / CC, c = gid % CC;
    const float* hf = hout + (size_t)row * HH;
    const float* hb = hout + (size_t)ROWS * HH + (size_t)row * HH;
    float ssum = b_out[c];
    for (int k = 0; k < HH; ++k) ssum += hf[k] * w_out[k * CC + c];
    for (int k = 0; k < HH; ++k) ssum += hb[k] * w_out[(HH + k) * CC + c];
    em[gid] = ssum;
}

// ---------------------------------------------------------------------------
// 7) valid mask -> d_out second half (float 0/1)
// ---------------------------------------------------------------------------
__global__ void k_valid(const int* __restrict__ len2, float* __restrict__ out) {
    int gid = blockIdx.x * 256 + threadIdx.x;
    if (gid >= ROWS) return;
    int b = gid / TP, t = gid % TP;
    out[ROWS + gid] = (t < len2[b]) ? 1.f : 0.f;
}

// ---------------------------------------------------------------------------
// 8) Viterbi decode, one lane per batch element (C=5)
// ---------------------------------------------------------------------------
__global__ void k_viterbi(const float* __restrict__ em, const int* __restrict__ len2,
                          const float* __restrict__ start, const float* __restrict__ endv,
                          const float* __restrict__ trans,
                          int* __restrict__ hist, float* __restrict__ out_tags) {
    int b = threadIdx.x;
    if (b >= BB) return;
    float score[CC];
#pragma unroll
    for (int c = 0; c < CC; ++c) score[c] = start[c] + em[(b * TP + 0) * CC + c];

    for (int t = 1; t < TP; ++t) {
        bool m = t < len2[b];
        float ns[CC]; int idx[CC];
#pragma unroll
        for (int c = 0; c < CC; ++c) {
            float best = -1e30f; int bi = 0;
#pragma unroll
            for (int p = 0; p < CC; ++p) {
                float v = score[p] + trans[p * CC + c];
                if (v > best) { best = v; bi = p; }
            }
            ns[c] = best + em[(b * TP + t) * CC + c];
            idx[c] = bi;
        }
#pragma unroll
        for (int c = 0; c < CC; ++c) {
            hist[((t - 1) * BB + b) * CC + c] = m ? idx[c] : c;
            if (m) score[c] = ns[c];
        }
    }
    float best = -1e30f; int last = 0;
#pragma unroll
    for (int c = 0; c < CC; ++c) {
        float v = score[c] + endv[c];
        if (v > best) { best = v; last = c; }
    }
    out_tags[b * TP + (TP - 1)] = (float)last;
    int tag = last;
    for (int t = TP - 1; t >= 1; --t) {
        tag = hist[((t - 1) * BB + b) * CC + tag];
        out_tags[b * TP + t - 1] = (float)tag;
    }
}

// ---------------------------------------------------------------------------
// host orchestration
// ---------------------------------------------------------------------------
extern "C" void kernel_launch(void* const* d_in, const int* in_sizes, int n_in,
                              void* d_out, int out_size, void* d_ws, size_t ws_size,
                              hipStream_t stream) {
    const float* emb    = (const float*)d_in[0];
    const int*   amask  = (const int*)d_in[1];
    const float* h0     = (const float*)d_in[2];
    const float* c0     = (const float*)d_in[3];
    const float* w_ih_f = (const float*)d_in[4];
    const float* w_hh_f = (const float*)d_in[5];
    const float* b_ih_f = (const float*)d_in[6];
    const float* b_hh_f = (const float*)d_in[7];
    const float* w_ih_b = (const float*)d_in[8];
    const float* w_hh_b = (const float*)d_in[9];
    const float* b_ih_b = (const float*)d_in[10];
    const float* b_hh_b = (const float*)d_in[11];
    const float* w_out  = (const float*)d_in[12];
    const float* b_out  = (const float*)d_in[13];
    const float* crf_s  = (const float*)d_in[14];
    const float* crf_e  = (const float*)d_in[15];
    const float* crf_t  = (const float*)d_in[16];
    float* out = (float*)d_out;

    // workspace carve-up (256B aligned)
    size_t off = 0;
    auto carve = [&](size_t bytes) -> void* {
        off = (off + 255) & ~(size_t)255;
        void* p = (char*)d_ws + off;
        off += bytes;
        return p;
    };
    int*            len2   = (int*)carve(BB * sizeof(int));
    unsigned short* xb     = (unsigned short*)carve((size_t)ROWS * DD * 2);
    unsigned short* wih_bf = (unsigned short*)carve((size_t)2 * G4H * DD * 2);
    unsigned short* whh_bf = (unsigned short*)carve((size_t)2 * G4H * HH * 2);
    float*          bias   = (float*)carve((size_t)2 * G4H * 4);
    float*          xg     = (float*)carve((size_t)2 * ROWS * G4H * 4);
    float*          hout   = (float*)carve((size_t)2 * ROWS * HH * 4);
    float*          em     = (float*)carve((size_t)ROWS * CC * 4);
    int*            hist   = (int*)carve((size_t)(TP - 1) * BB * CC * 4);
    (void)ws_size; (void)n_in; (void)in_sizes; (void)out_size;

    // large dynamic LDS for the recurrent kernel (128KB g + 16KB hA)
    const int lstmLds = 16 * G4H * 4 + 16 * HH * 2;
    hipFuncSetAttribute((const void*)k_lstm,
                        hipFuncAttributeMaxDynamicSharedMemorySize, lstmLds);

    k_lengths<<<BB, 256, 0, stream>>>(amask, len2);

    {
        long long tot = (long long)ROWS * DD;
        int blocks = (int)((tot + 255) / 256);
        k_pack<<<blocks, 256, 0, stream>>>(emb, len2, xb);
    }

    {
        int n = G4H * DD;
        k_cvt_bf16<<<(n + 255) / 256, 256, 0, stream>>>(w_ih_f, wih_bf, n);
        k_cvt_bf16<<<(n + 255) / 256, 256, 0, stream>>>(w_ih_b, wih_bf + (size_t)G4H * DD, n);
        int m = G4H * HH;
        k_cvt_bf16<<<(m + 255) / 256, 256, 0, stream>>>(w_hh_f, whh_bf, m);
        k_cvt_bf16<<<(m + 255) / 256, 256, 0, stream>>>(w_hh_b, whh_bf + (size_t)G4H * HH, m);
        k_bias<<<(2 * G4H + 255) / 256, 256, 0, stream>>>(b_ih_f, b_hh_f, b_ih_b, b_hh_b, bias);
    }

    {
        dim3 grid(ROWS / 16, G4H / 512, 2);
        int lds = 16 * (DD / 2) * 4;  // 40KB A tile
        k_xg_gemm<<<grid, 256, lds, stream>>>(xb, wih_bf, bias, xg);
    }

    k_lstm<<<2, 1024, lstmLds, stream>>>(xg, whh_bf, h0, c0, hout);

    k_emissions<<<(ROWS * CC + 255) / 256, 256, 0, stream>>>(hout, w_out, b_out, em);
    k_valid<<<(ROWS + 255) / 256, 256, 0, stream>>>(len2, out);
    k_viterbi<<<1, 32, 0, stream>>>(em, len2, crf_s, crf_e, crf_t, hist, out);
}
